// GraphMambaLayer_19550691131592
// MI455X (gfx1250) — compile-verified, matrix operand order checked by reading, not run on hardware
//
#include <hip/hip_runtime.h>
#include <math.h>

typedef __bf16 bf16_t;
typedef __attribute__((ext_vector_type(16))) __bf16 bf16x16;
typedef __attribute__((ext_vector_type(8)))  float  f32x8;

#define NLAYERS 2

// ---------------- small device helpers ----------------

__device__ __forceinline__ bf16_t f2bf(float f) {
  union { float f; unsigned u; } in; in.f = f;
  unsigned u = in.u;
  u += 0x7FFFu + ((u >> 16) & 1u);          // round-to-nearest-even
  union { unsigned short s; bf16_t b; } out;
  out.s = (unsigned short)(u >> 16);
  return out.b;
}

__device__ __forceinline__ float geluf(float x) {
  return 0.5f * x * (1.0f + erff(x * 0.70710678118654752f));
}
__device__ __forceinline__ float sigm(float x) {
  return 1.0f / (1.0f + __expf(-x));
}

// gfx1250: async global->LDS DMA (ASYNCcnt), bypasses VGPRs entirely.
__device__ __forceinline__ void async_load_lds_b128(unsigned lds_addr, const void* gaddr) {
  asm volatile("global_load_async_to_lds_b128 %0, %1, off"
               :: "v"(lds_addr), "v"((unsigned long long)gaddr) : "memory");
}
__device__ __forceinline__ void wait_async_zero() {
  asm volatile("s_wait_asynccnt 0x0" ::: "memory");
}
__device__ __forceinline__ unsigned lds_addr_of(const void* p) {
  // generic address of __shared__ has LDS byte offset in low 32 bits (ISA 10.2)
  return (unsigned)(unsigned long long)p;
}

// block = 256 threads (8 wave32)
__device__ float blockReduceSum256(float v, float* sbuf) {
  int lane = threadIdx.x & 31;
  int w    = threadIdx.x >> 5;
  #pragma unroll
  for (int o = 16; o > 0; o >>= 1) v += __shfl_down(v, o, 32);
  if (lane == 0) sbuf[w] = v;
  __syncthreads();
  if (threadIdx.x == 0) {
    float t = 0.f;
    #pragma unroll
    for (int i = 0; i < 8; ++i) t += sbuf[i];
    sbuf[0] = t;
  }
  __syncthreads();
  float r = sbuf[0];
  __syncthreads();
  return r;
}

// ---------------- bf16 WMMA GEMM ----------------
// C[M,N] = epilogue( A[M,K](bf16) * BT[N,K]^T(bf16) + bias[N] )
// tiles: 128x128x32, 256 threads = 8 waves in 2(M) x 4(N), each wave 64x32 = 4x2 WMMA accs.
// K-tiles staged into LDS with global_load_async_to_lds_b128, double buffered.

#define TBM 128
#define TBN 128
#define TBK 32
#define LDSW 40   // halfs per LDS row (32 data + 8 pad -> conflict-free b128 reads)

template<bool OUTBF, bool DOGELU>
__global__ __launch_bounds__(256) void gemm_bf16_wmma(
    const bf16_t* __restrict__ A, int lda,
    const bf16_t* __restrict__ BT,          // [N, K] row-major (weight transposed)
    const float*  __restrict__ bias,        // [N]
    const float*  __restrict__ addsrc, int add_ld,   // optional residual add
    const float*  __restrict__ km,          // optional per-row scale
    void* __restrict__ Cout, int ldc, int K)
{
  __shared__ __align__(16) bf16_t As[2][TBM * LDSW];
  __shared__ __align__(16) bf16_t Bs[2][TBN * LDSW];

  const int tid  = threadIdx.x;
  const int wave = tid >> 5;
  const int lane = tid & 31;
  const int wm   = wave & 1;    // 0..1
  const int wn   = wave >> 1;   // 0..3
  const int m16  = lane & 15;
  const int h    = lane >> 4;

  const long long m0 = (long long)blockIdx.y * TBM;
  const long long n0 = (long long)blockIdx.x * TBN;

  f32x8 acc[4][2];
  #pragma unroll
  for (int i = 0; i < 4; ++i)
    #pragma unroll
    for (int j = 0; j < 2; ++j)
      #pragma unroll
      for (int r = 0; r < 8; ++r) acc[i][j][r] = 0.f;

  // per-thread staging geometry (invariant over k)
  const int seg0 = tid;                  // 0..255
  const int row0 = seg0 >> 2;            // 0..63
  const int cs0  = (seg0 & 3) << 3;      // halfs
  const int seg1 = tid + 256;            // 256..511
  const int row1 = seg1 >> 2;            // 64..127
  const int cs1  = (seg1 & 3) << 3;

  auto stage = [&](int kt, int buf) {
    const long long k0 = (long long)kt * TBK;
    async_load_lds_b128(lds_addr_of(&As[buf][row0 * LDSW + cs0]),
                        A  + (m0 + row0) * (long long)lda + k0 + cs0);
    async_load_lds_b128(lds_addr_of(&Bs[buf][row0 * LDSW + cs0]),
                        BT + (n0 + row0) * (long long)K   + k0 + cs0);
    async_load_lds_b128(lds_addr_of(&As[buf][row1 * LDSW + cs1]),
                        A  + (m0 + row1) * (long long)lda + k0 + cs1);
    async_load_lds_b128(lds_addr_of(&Bs[buf][row1 * LDSW + cs1]),
                        BT + (n0 + row1) * (long long)K   + k0 + cs1);
  };

  auto compute = [&](int buf) {
    bf16x16 af[4], bfr[2];
    #pragma unroll
    for (int mi = 0; mi < 4; ++mi) {
      const bf16_t* p = &As[buf][(wm * 64 + mi * 16 + m16) * LDSW];
      union { bf16x16 v; uint4 q[2]; } u;
      u.q[0] = *(const uint4*)(p + h * 8);
      u.q[1] = *(const uint4*)(p + 16 + h * 8);
      af[mi] = u.v;
    }
    #pragma unroll
    for (int ni = 0; ni < 2; ++ni) {
      const bf16_t* p = &Bs[buf][(wn * 32 + ni * 16 + m16) * LDSW];
      union { bf16x16 v; uint4 q[2]; } u;
      u.q[0] = *(const uint4*)(p + h * 8);
      u.q[1] = *(const uint4*)(p + 16 + h * 8);
      bfr[ni] = u.v;
    }
    #pragma unroll
    for (int mi = 0; mi < 4; ++mi)
      #pragma unroll
      for (int ni = 0; ni < 2; ++ni)
        acc[mi][ni] = __builtin_amdgcn_wmma_f32_16x16x32_bf16(
            false, af[mi], false, bfr[ni], (short)0, acc[mi][ni], false, false);
  };

  const int KT = K / TBK;
  stage(0, 0);
  wait_async_zero();
  __syncthreads();
  for (int kt = 0; kt < KT; ++kt) {
    int cur = kt & 1;
    if (kt + 1 < KT) stage(kt + 1, cur ^ 1);   // async DMA overlaps WMMA below
    compute(cur);
    wait_async_zero();
    __syncthreads();
  }

  // epilogue: C/D layout -> VGPR r, lane l: M = 8*(l/16)+r, N = l%16 within 16x16 tile
  #pragma unroll
  for (int mi = 0; mi < 4; ++mi) {
    #pragma unroll
    for (int ni = 0; ni < 2; ++ni) {
      long long n  = n0 + wn * 32 + ni * 16 + m16;
      long long mb = m0 + wm * 64 + mi * 16 + h * 8;
      float bv = bias[n];
      #pragma unroll
      for (int r = 0; r < 8; ++r) {
        long long m = mb + r;
        float v = acc[mi][ni][r] + bv;
        if (DOGELU) {
          v = geluf(v);
        } else {
          if (addsrc) v += addsrc[m * (long long)add_ld + n];
          if (km)     v *= km[m];
        }
        if (OUTBF) ((bf16_t*)Cout)[m * (long long)ldc + n] = f2bf(v);
        else       ((float*) Cout)[m * (long long)ldc + n] = v;
      }
    }
  }
}

// ---------------- transpose + f32->bf16 convert (weights) ----------------
// in[K,N] f32 -> out[N,Kpad] bf16, zero-padded rows K..Kpad
__global__ __launch_bounds__(256) void transpose_cvt(
    const float* __restrict__ in, bf16_t* __restrict__ out,
    int K, int N, int Kpad)
{
  __shared__ float tile[32][33];
  int k0 = blockIdx.x * 32, n0 = blockIdx.y * 32;
  int tx = threadIdx.x & 31, ty = threadIdx.x >> 5;
  #pragma unroll
  for (int j = 0; j < 4; ++j) {
    int k = k0 + ty + j * 8;
    int n = n0 + tx;
    tile[ty + j * 8][tx] = (k < K && n < N) ? in[(long long)k * N + n] : 0.f;
  }
  __syncthreads();
  #pragma unroll
  for (int j = 0; j < 4; ++j) {
    int n = n0 + ty + j * 8;
    int k = k0 + tx;
    if (n < N && k < Kpad)
      out[(long long)n * Kpad + k] = f2bf(tile[tx][ty + j * 8]);
  }
}

// ---------------- attn-map stats + sinusoidal encoding ----------------
__global__ __launch_bounds__(256) void stats_enc_kernel(
    const float* __restrict__ attn, const int* __restrict__ Hp,
    const int* __restrict__ Wp, int Npix, bf16_t* __restrict__ enc, int Kpad)
{
  __shared__ float sbuf[8];
  long long row = blockIdx.x;
  int H = *Hp, W = *Wp;
  float iy = 1.f / (float)((H - 1) > 1 ? (H - 1) : 1);
  float ix = 1.f / (float)((W - 1) > 1 ? (W - 1) : 1);
  const float* a = attn + row * (long long)Npix;
  float S = 0.f, Sy = 0.f, Sx = 0.f, Sq = 0.f;
  int y = threadIdx.x / W, x = threadIdx.x % W;
  int dy = 256 / W, dx = 256 % W;
  for (int n = threadIdx.x; n < Npix; n += 256) {
    float v = a[n];
    float gy = (float)y * iy;
    float gx = (float)x * ix;
    S += v; Sy += v * gy; Sx += v * gx; Sq += v * (gy * gy + gx * gx);
    y += dy; x += dx;
    if (x >= W) { x -= W; y += 1; }
  }
  S  = blockReduceSum256(S,  sbuf);
  Sy = blockReduceSum256(Sy, sbuf);
  Sx = blockReduceSum256(Sx, sbuf);
  Sq = blockReduceSum256(Sq, sbuf);
  float inv = 1.f / (S + 1e-8f);
  float cy = Sy * inv, cx = Sx * inv;
  float sp2 = (Sq - 2.f * cy * Sy - 2.f * cx * Sx + (cy * cy + cx * cx) * S) * inv;
  float sp = sqrtf(fmaxf(sp2, 0.f));
  if ((int)threadIdx.x < Kpad) {
    int j = threadIdx.x;
    const float c = 0.28782313662425572f;  // ln(10000)/32
    float val = 0.f;
    if (j < 32)        val = sinf(cy * expf(-(float)j        * c));
    else if (j < 64)   val = cosf(cy * expf(-(float)(j - 32) * c));
    else if (j < 96)   val = sinf(cx * expf(-(float)(j - 64) * c));
    else if (j < 128)  val = cosf(cx * expf(-(float)(j - 96) * c));
    else if (j == 128) val = sp;
    enc[row * (long long)Kpad + j] = f2bf(val);
  }
}

// ---------------- LN of s with two parameter sets -> two bf16 outputs ----------------
__global__ __launch_bounds__(256) void ln_dual_kernel(
    const float* __restrict__ S, int D,
    const float* __restrict__ g1, const float* __restrict__ b1,
    const float* __restrict__ g2, const float* __restrict__ b2,
    bf16_t* __restrict__ o1, bf16_t* __restrict__ o2)
{
  __shared__ float sbuf[8];
  long long row = blockIdx.x;
  const float* x = S + row * (long long)D;
  float vals[8];
  int c = D >> 8;
  float s = 0.f;
  for (int i = 0; i < c; ++i) { vals[i] = x[threadIdx.x + i * 256]; s += vals[i]; }
  float mean = blockReduceSum256(s, sbuf) / (float)D;
  float vs = 0.f;
  for (int i = 0; i < c; ++i) { float d = vals[i] - mean; vs += d * d; }
  float rs = rsqrtf(blockReduceSum256(vs, sbuf) / (float)D + 1e-5f);
  for (int i = 0; i < c; ++i) {
    int j = threadIdx.x + i * 256;
    float nh = (vals[i] - mean) * rs;
    o1[row * (long long)D + j] = f2bf(nh * g1[j] + b1[j]);
    o2[row * (long long)D + j] = f2bf(nh * g2[j] + b2[j]);
  }
}

// ---------------- silu(z)*sigmoid(gate) then LN -> bf16 ----------------
__global__ __launch_bounds__(256) void gate_ln_kernel(
    const float* __restrict__ ZI, int DI,
    const float* __restrict__ ig, const float* __restrict__ ib,
    bf16_t* __restrict__ U)
{
  __shared__ float sbuf[8];
  long long row = blockIdx.x;
  const float* z  = ZI + row * (long long)(2 * DI);
  const float* gt = z + DI;
  float vals[8];
  int c = DI >> 8;
  float s = 0.f;
  for (int i = 0; i < c; ++i) {
    int j = threadIdx.x + i * 256;
    float zz = z[j], gg = gt[j];
    float v = zz * sigm(zz) * sigm(gg);
    vals[i] = v; s += v;
  }
  float mean = blockReduceSum256(s, sbuf) / (float)DI;
  float vs = 0.f;
  for (int i = 0; i < c; ++i) { float d = vals[i] - mean; vs += d * d; }
  float rs = rsqrtf(blockReduceSum256(vs, sbuf) / (float)DI + 1e-5f);
  for (int i = 0; i < c; ++i) {
    int j = threadIdx.x + i * 256;
    U[row * (long long)DI + j] = f2bf((vals[i] - mean) * rs * ig[j] + ib[j]);
  }
}

// ---------------- s = LN(s + m)*g1+b1 ; X = bf16(LN(s)*g2+b2) ----------------
__global__ __launch_bounds__(256) void add_ln_ln_kernel(
    float* __restrict__ S, const float* __restrict__ Madd, int D,
    const float* __restrict__ g1, const float* __restrict__ b1,
    const float* __restrict__ g2, const float* __restrict__ b2,
    bf16_t* __restrict__ X)
{
  __shared__ float sbuf[8];
  long long row = blockIdx.x;
  float* x = S + row * (long long)D;
  const float* mm = Madd + row * (long long)D;
  float vals[8];
  int c = D >> 8;
  float s = 0.f;
  for (int i = 0; i < c; ++i) {
    int j = threadIdx.x + i * 256;
    vals[i] = x[j] + mm[j]; s += vals[i];
  }
  float mean = blockReduceSum256(s, sbuf) / (float)D;
  float vs = 0.f;
  for (int i = 0; i < c; ++i) { float d = vals[i] - mean; vs += d * d; }
  float rs = rsqrtf(blockReduceSum256(vs, sbuf) / (float)D + 1e-5f);
  float s2 = 0.f;
  for (int i = 0; i < c; ++i) {
    int j = threadIdx.x + i * 256;
    float y = (vals[i] - mean) * rs * g1[j] + b1[j];
    vals[i] = y; s2 += y;
  }
  float mean2 = blockReduceSum256(s2, sbuf) / (float)D;
  float vs2 = 0.f;
  for (int i = 0; i < c; ++i) { float d = vals[i] - mean2; vs2 += d * d; }
  float rs2 = rsqrtf(blockReduceSum256(vs2, sbuf) / (float)D + 1e-5f);
  for (int i = 0; i < c; ++i) {
    int j = threadIdx.x + i * 256;
    x[j] = vals[i];
    X[row * (long long)D + j] = f2bf((vals[i] - mean2) * rs2 * g2[j] + b2[j]);
  }
}

// ---------------- host launcher ----------------

extern "C" void kernel_launch(void* const* d_in, const int* in_sizes, int n_in,
                              void* d_out, int out_size, void* d_ws, size_t ws_size,
                              hipStream_t stream) {
  const float* s_in   = (const float*)d_in[0];
  const float* attn   = (const float*)d_in[1];
  const float* km     = (const float*)d_in[2];
  const float* se_w1  = (const float*)d_in[3];
  const float* se_b1  = (const float*)d_in[4];
  const float* se_w2  = (const float*)d_in[5];
  const float* se_b2  = (const float*)d_in[6];
  const float* f_ng   = (const float*)d_in[7];
  const float* f_nb   = (const float*)d_in[8];
  const float* f_piw  = (const float*)d_in[9];
  const float* f_pib  = (const float*)d_in[10];
  const float* f_pow  = (const float*)d_in[11];
  const float* f_pob  = (const float*)d_in[12];
  const float* f_ig   = (const float*)d_in[13];
  const float* f_ib   = (const float*)d_in[14];
  const float* b_ng   = (const float*)d_in[15];
  const float* b_nb   = (const float*)d_in[16];
  const float* b_piw  = (const float*)d_in[17];
  const float* b_pib  = (const float*)d_in[18];
  const float* b_pow  = (const float*)d_in[19];
  const float* b_pob  = (const float*)d_in[20];
  const float* b_ig   = (const float*)d_in[21];
  const float* b_ib   = (const float*)d_in[22];
  const float* mrg_w  = (const float*)d_in[23];
  const float* mrg_b  = (const float*)d_in[24];
  const float* nrm_g  = (const float*)d_in[25];
  const float* nrm_b  = (const float*)d_in[26];
  const float* ffn_ng = (const float*)d_in[27];
  const float* ffn_nb = (const float*)d_in[28];
  const float* ffn_w1 = (const float*)d_in[29];
  const float* ffn_b1 = (const float*)d_in[30];
  const float* ffn_w2 = (const float*)d_in[31];
  const float* ffn_b2 = (const float*)d_in[32];
  const int*   Hp     = (const int*)d_in[33];
  const int*   Wp     = (const int*)d_in[34];
  (void)n_in; (void)out_size; (void)ws_size;

  const int ENCC = 129;                       // 2*ENC + 1
  const int D    = in_sizes[3] / ENCC;
  const long long BK = (long long)in_sizes[0] / D;
  const int Npix = (int)((long long)in_sizes[1] / BK);
  const int DI2  = in_sizes[10] / NLAYERS;    // 2*DI
  const int DI   = DI2 / 2;
  const int DH   = in_sizes[30] / NLAYERS;    // 4*D
  const int D2   = 2 * D;
  const int Kpad = ((ENCC + 31) / 32) * 32;   // 160

  // ---- workspace carve ----
  char* base = (char*)d_ws;
  size_t off = 0;
  auto alloc = [&](size_t bytes) -> char* {
    off = (off + 255) & ~(size_t)255;
    char* p = base + off;
    off += bytes;
    return p;
  };
  bf16_t* w_se1 = (bf16_t*)alloc((size_t)D * Kpad * 2);
  bf16_t* w_se2 = (bf16_t*)alloc((size_t)D * D * 2);
  bf16_t* w_piw[2][NLAYERS];
  bf16_t* w_pow[2][NLAYERS];
  bf16_t* w_mrg[NLAYERS];
  bf16_t* w_ff1[NLAYERS];
  bf16_t* w_ff2[NLAYERS];
  for (int d = 0; d < 2; ++d)
    for (int l = 0; l < NLAYERS; ++l) w_piw[d][l] = (bf16_t*)alloc((size_t)DI2 * D * 2);
  for (int d = 0; d < 2; ++d)
    for (int l = 0; l < NLAYERS; ++l) w_pow[d][l] = (bf16_t*)alloc((size_t)D * DI * 2);
  for (int l = 0; l < NLAYERS; ++l) w_mrg[l] = (bf16_t*)alloc((size_t)D * D2 * 2);
  for (int l = 0; l < NLAYERS; ++l) w_ff1[l] = (bf16_t*)alloc((size_t)DH * D * 2);
  for (int l = 0; l < NLAYERS; ++l) w_ff2[l] = (bf16_t*)alloc((size_t)D * DH * 2);
  bf16_t* enc  = (bf16_t*)alloc((size_t)BK * Kpad * 2);
  bf16_t* xf   = (bf16_t*)alloc((size_t)BK * D * 2);
  bf16_t* xb   = (bf16_t*)alloc((size_t)BK * D * 2);
  bf16_t* ubuf = (bf16_t*)alloc((size_t)BK * DI * 2);
  bf16_t* fb   = (bf16_t*)alloc((size_t)BK * D2 * 2);
  float*  mbuf = (float*) alloc((size_t)BK * D * 4);
  float*  zbuf = (float*) alloc((size_t)BK * DI2 * 4);  // also reused as bf16 g1 [BK, DH]

  dim3 tb(256);
  auto tgrid = [](int Kp, int N) { return dim3((unsigned)((Kp + 31) / 32), (unsigned)((N + 31) / 32)); };

  // ---- weight transpose + bf16 convert ----
  transpose_cvt<<<tgrid(Kpad, D), tb, 0, stream>>>(se_w1, w_se1, ENCC, D, Kpad);
  transpose_cvt<<<tgrid(D, D),    tb, 0, stream>>>(se_w2, w_se2, D, D, D);
  for (int l = 0; l < NLAYERS; ++l) {
    transpose_cvt<<<tgrid(D, DI2), tb, 0, stream>>>(f_piw + (size_t)l * D * DI2, w_piw[0][l], D, DI2, D);
    transpose_cvt<<<tgrid(D, DI2), tb, 0, stream>>>(b_piw + (size_t)l * D * DI2, w_piw[1][l], D, DI2, D);
    transpose_cvt<<<tgrid(DI, D),  tb, 0, stream>>>(f_pow + (size_t)l * DI * D,  w_pow[0][l], DI, D, DI);
    transpose_cvt<<<tgrid(DI, D),  tb, 0, stream>>>(b_pow + (size_t)l * DI * D,  w_pow[1][l], DI, D, DI);
    transpose_cvt<<<tgrid(D2, D),  tb, 0, stream>>>(mrg_w + (size_t)l * D2 * D,  w_mrg[l],    D2, D, D2);
    transpose_cvt<<<tgrid(D, DH),  tb, 0, stream>>>(ffn_w1 + (size_t)l * D * DH, w_ff1[l],    D, DH, D);
    transpose_cvt<<<tgrid(DH, D),  tb, 0, stream>>>(ffn_w2 + (size_t)l * DH * D, w_ff2[l],    DH, D, DH);
  }

  // ---- attn stats + sinusoidal encoding ----
  stats_enc_kernel<<<dim3((unsigned)BK), tb, 0, stream>>>(attn, Hp, Wp, Npix, enc, Kpad);

  auto gemm = [&](const bf16_t* A, int lda, const bf16_t* BT, const float* bias,
                  const float* add, int add_ld, const float* kmv,
                  void* C, int ldc, long long M, int N, int K, bool outbf, bool dogelu) {
    dim3 g((unsigned)(N / TBN), (unsigned)(M / TBM));
    if (dogelu)
      gemm_bf16_wmma<true, true><<<g, tb, 0, stream>>>(A, lda, BT, bias, nullptr, 0, nullptr, C, ldc, K);
    else if (outbf)
      gemm_bf16_wmma<true, false><<<g, tb, 0, stream>>>(A, lda, BT, bias, add, add_ld, kmv, C, ldc, K);
    else
      gemm_bf16_wmma<false, false><<<g, tb, 0, stream>>>(A, lda, BT, bias, add, add_ld, kmv, C, ldc, K);
  };

  float* S = (float*)d_out;   // f32 master copy of s, fully rewritten below

  // s = s_in + gelu(enc @ se_w1 + b1) @ se_w2 + b2
  gemm(enc, Kpad, w_se1, se_b1, nullptr, 0, nullptr, xf, D, BK, D, Kpad, true, true);
  gemm(xf,  D,    w_se2, se_b2, s_in,    D, nullptr, S,  D, BK, D, D,    false, false);

  // NOTE: order/rv permutation and flip cancel exactly (all layer ops are per-token).
  for (int l = 0; l < NLAYERS; ++l) {
    ln_dual_kernel<<<dim3((unsigned)BK), tb, 0, stream>>>(
        S, D, f_ng + (size_t)l * D, f_nb + (size_t)l * D,
              b_ng + (size_t)l * D, b_nb + (size_t)l * D, xf, xb);
    // forward SSM core -> fb[:, 0:D]
    gemm(xf, D, w_piw[0][l], f_pib + (size_t)l * DI2, nullptr, 0, nullptr, zbuf, DI2, BK, DI2, D, false, false);
    gate_ln_kernel<<<dim3((unsigned)BK), tb, 0, stream>>>(zbuf, DI, f_ig + (size_t)l * DI, f_ib + (size_t)l * DI, ubuf);
    gemm(ubuf, DI, w_pow[0][l], f_pob + (size_t)l * D, S, D, nullptr, fb, D2, BK, D, DI, true, false);
    // backward SSM core -> fb[:, D:2D]
    gemm(xb, D, w_piw[1][l], b_pib + (size_t)l * DI2, nullptr, 0, nullptr, zbuf, DI2, BK, DI2, D, false, false);
    gate_ln_kernel<<<dim3((unsigned)BK), tb, 0, stream>>>(zbuf, DI, b_ig + (size_t)l * DI, b_ib + (size_t)l * DI, ubuf);
    gemm(ubuf, DI, w_pow[1][l], b_pob + (size_t)l * D, S, D, nullptr, fb + D, D2, BK, D, DI, true, false);
    // merge: m = [f, bb] @ mrg_w + mrg_b
    gemm(fb, D2, w_mrg[l], mrg_b + (size_t)l * D, nullptr, 0, nullptr, mbuf, D, BK, D, D2, false, false);
    // s = LN(s + m); xf = bf16(LN(s, ffn))
    add_ln_ln_kernel<<<dim3((unsigned)BK), tb, 0, stream>>>(
        S, mbuf, D, nrm_g + (size_t)l * D, nrm_b + (size_t)l * D,
                    ffn_ng + (size_t)l * D, ffn_nb + (size_t)l * D, xf);
    // FFN: s = (s + gelu(xf @ w1 + b1) @ w2 + b2) * km
    bf16_t* g1 = (bf16_t*)zbuf;
    gemm(xf, D, w_ff1[l], ffn_b1 + (size_t)l * DH, nullptr, 0, nullptr, g1, DH, BK, DH, D, true, true);
    gemm(g1, DH, w_ff2[l], ffn_b2 + (size_t)l * D, S, D, km, S, D, BK, D, DH, false, false);
  }
}
